// QuestionDescriminatorModel_52810917872158
// MI455X (gfx1250) — matrix operand order repaired and checked by reading, not compiled, not run on hardware
//
#include <hip/hip_runtime.h>

typedef unsigned short u16;
typedef unsigned int   u32;
typedef unsigned long long u64;
typedef __attribute__((ext_vector_type(16))) __bf16 v16bf;
typedef __attribute__((ext_vector_type(8)))  float  v8f;
typedef __attribute__((ext_vector_type(4)))  unsigned int v4u;
typedef __attribute__((ext_vector_type(8)))  int v8i;
typedef __attribute__((ext_vector_type(4)))  int v4i;

#define Bsz 2
#define Ssz 4096
#define Dsz 768
#define Hsz 12
#define DHsz 64
#define NBsz 256
#define FFsz 3072
#define ROWS (Bsz*Ssz)          /* 8192 */
#define QKVW (3*Dsz)            /* 2304 */
#define SCALE 0.125f

union Frag { uint4 q[2]; v16bf v; };

__device__ __forceinline__ u16 f2bf(float f) {
  u32 u = __float_as_uint(f);
  u += 0x7fffu + ((u >> 16) & 1u);
  return (u16)(u >> 16);
}

__device__ __forceinline__ v8f v8f_zero() {
  v8f z = {0.f,0.f,0.f,0.f,0.f,0.f,0.f,0.f};
  return z;
}

__device__ __forceinline__ v8f wmma_bf16(const Frag& a, const Frag& b, v8f c) {
  return __builtin_amdgcn_wmma_f32_16x16x32_bf16(false, a.v, false, b.v,
                                                 (short)0, c, false, false);
}

__device__ __forceinline__ float gelu_tanh(float v) {
  float c = 0.7978845608028654f * (v + 0.044715f * v * v * v);
  return 0.5f * v * (1.0f + tanhf(c));
}

// CDNA5 async global->LDS copy (ASYNCcnt-tracked, no VGPR bounce).
__device__ __forceinline__ void async_b128(u32 lds_off, const u16* g) {
  asm volatile("global_load_async_to_lds_b128 %0, %1, off"
               :: "v"(lds_off), "v"(g) : "memory");
}
__device__ __forceinline__ void wait_async0() {
  asm volatile("s_wait_asynccnt 0x0" ::: "memory");
}

// CDNA5 Tensor Data Mover: DMA a 2D tile (128 rows x 32 bf16, row stride K elems)
// from global to LDS, inserting 16B padding per 64B row (-> 80B LDS row stride).
// Descriptor per CDNA5 ISA; TENSORcnt-tracked.
// This toolchain exposes the 6-arg builtin: (g0, g1, g2, g3, g1b, cpol).
__device__ __forceinline__ void tdm_load_tile(u32 lds_byte, const u16* gsrc, int K) {
  u64 ga = (u64)(size_t)gsrc;
  v4u g0;
  g0[0] = 1u;                                       // count=1, user descriptor
  g0[1] = lds_byte;                                 // lds_addr
  g0[2] = (u32)(ga & 0xffffffffull);                // global_addr[31:0]
  g0[3] = (u32)((ga >> 32) & 0x1ffffffull)          // global_addr[56:32]
          | (2u << 30);                             // type=2 ("image")
  u64 td0 = (u64)(u32)K;                            // tensor_dim0 (elems per row)
  u64 td1 = 1ull << 20;                             // tensor_dim1 (rows, large)
  u64 st0 = (u64)(u32)K;                            // tensor_dim0_stride (elems)
  v8i g1;
  g1[0] = (int)((1u << 16)                          // data_size = 2 bytes
              | (1u << 20)                          // pad_enable
              | (3u << 22)                          // pad_interval: 16 DWORDs
              | (3u << 25));                        // pad_amount:   4 DWORDs
  g1[1] = (int)((td0 & 0xffffull) << 16);           // tensor_dim0[15:0]
  g1[2] = (int)((td0 >> 16) | ((td1 & 0xffffull) << 16));
  g1[3] = (int)((td1 >> 16) | (32ull << 16));       // tile_dim0 = 32
  g1[4] = (int)128;                                 // tile_dim1 = 128 (tile_dim2=0)
  g1[5] = (int)(st0 & 0xffffffffull);
  g1[6] = (int)(st0 >> 32);
  g1[7] = 0;
  v4i z4 = {0, 0, 0, 0};
  v8i z8 = {0, 0, 0, 0, 0, 0, 0, 0};
  __builtin_amdgcn_tensor_load_to_lds(g0, g1, z4, z4, z8, 0);
}

// ---------------------------------------------------------------- embedding
__global__ __launch_bounds__(256) void embed_kernel(const int* __restrict__ ids,
                                                    const float* __restrict__ emb,
                                                    float* __restrict__ xf,
                                                    u16* __restrict__ xb) {
  int row = blockIdx.x;
  int id  = ids[row];
  const float* e = emb + (size_t)id * Dsz;
  for (int j = threadIdx.x; j < Dsz; j += 256) {
    float v = e[j];
    xf[(size_t)row * Dsz + j] = v;
    xb[(size_t)row * Dsz + j] = f2bf(v);
  }
}

// ---------------------------------------------------------------- weight transpose + bf16
// W (L,K,N) f32 -> Wt (L,N,K) bf16 ; one-time, weights stay L2-resident.
__global__ __launch_bounds__(256) void transpose_f2bf(const float* __restrict__ W,
                                                      u16* __restrict__ Wt,
                                                      int K, int N) {
  __shared__ float tile[32][33];
  const int l = blockIdx.z;
  const float* Wl = W + (size_t)l * K * N;
  u16* Wtl = Wt + (size_t)l * K * N;
  const int k0 = blockIdx.y * 32, n0 = blockIdx.x * 32;
  const int tx = threadIdx.x & 31, ty = threadIdx.x >> 5;  // 32 x 8
  for (int i = ty; i < 32; i += 8)
    tile[i][tx] = Wl[(size_t)(k0 + i) * N + n0 + tx];
  __syncthreads();
  for (int i = ty; i < 32; i += 8)
    Wtl[(size_t)(n0 + i) * K + k0 + tx] = f2bf(tile[tx][i]);
}

// ---------------------------------------------------------------- WMMA GEMM
// C[M=8192 x N] = A[M x K](bf16) @ Wt[N x K](bf16, pre-transposed) + bias.
// Block tile 128x128, 8 waves (4Mx2N), wave tile 32x64 (2x4 wmma), K step 32.
// A tile: per-lane async global->LDS b128. B (weight) tile: one TDM descriptor
// issued by wave 0 (TENSORcnt). Both double-buffered; DMA overlaps WMMA.
#define ABUF (128*40*2)   /* bytes per LDS buffer */
__global__ __launch_bounds__(256) void gemm_bf16(const u16* __restrict__ A,
                                                 const u16* __restrict__ Wt,
                                                 const float* __restrict__ bias,
                                                 float* __restrict__ Cf,
                                                 u16* __restrict__ Cb,
                                                 int N, int K, int do_gelu) {
  __shared__ u16 As[2][128][40];   // [buf][m][k]
  __shared__ u16 Bs[2][128][40];   // [buf][n][k]
  const int bm = blockIdx.y * 128;
  const int bn = blockIdx.x * 128;
  const int tid  = threadIdx.x;
  const int wid  = tid >> 5;
  const int lane = tid & 31;
  const int half = lane >> 4, l15 = lane & 15;
  const int wm = (wid >> 1) * 32;   // 0,32,64,96
  const int wn = (wid & 1) * 64;    // 0,64

  v8f acc[2][4];
  #pragma unroll
  for (int i = 0; i < 2; ++i)
    #pragma unroll
    for (int j = 0; j < 4; ++j) acc[i][j] = v8f_zero();

  // A staging map: 256 threads cover 128 rows x 32 k, 16 elems (2 x b128) each
  const int sr  = tid >> 1;
  const int sof = (tid & 1) * 16;
  const u32 asA = (u32)(size_t)(&As[0][0][0]);
  const u32 asB = (u32)(size_t)(&Bs[0][0][0]);
  const u32 soff = (u32)((sr * 40 + sof) * 2);

  auto stageA = [&](int kt, int buf) {
    const u16* ga = A + (size_t)(bm + sr) * K + kt * 32 + sof;
    u32 da = asA + (u32)buf * ABUF + soff;
    async_b128(da,      ga);
    async_b128(da + 16, ga + 8);
  };
  auto stageB = [&](int kt, int buf) {   // wave-uniform; call from wave 0 only
    tdm_load_tile(asB + (u32)buf * ABUF, Wt + (size_t)bn * K + kt * 32, K);
  };

  auto compute = [&](int buf) {
    Frag a0, a1, bf[4];
    a0.q[0] = *(const uint4*)&As[buf][wm +  0 + l15][half * 8];
    a0.q[1] = *(const uint4*)&As[buf][wm +  0 + l15][half * 8 + 16];
    a1.q[0] = *(const uint4*)&As[buf][wm + 16 + l15][half * 8];
    a1.q[1] = *(const uint4*)&As[buf][wm + 16 + l15][half * 8 + 16];
    #pragma unroll
    for (int tj = 0; tj < 4; ++tj) {
      bf[tj].q[0] = *(const uint4*)&Bs[buf][wn + tj * 16 + l15][half * 8];
      bf[tj].q[1] = *(const uint4*)&Bs[buf][wn + tj * 16 + l15][half * 8 + 16];
    }
    #pragma unroll
    for (int tj = 0; tj < 4; ++tj) {
      acc[0][tj] = wmma_bf16(a0, bf[tj], acc[0][tj]);
      acc[1][tj] = wmma_bf16(a1, bf[tj], acc[1][tj]);
    }
  };

  const int nK = K >> 5;
  stageA(0, 0);
  if (wid == 0) stageB(0, 0);
  wait_async0();
  if (wid == 0) __builtin_amdgcn_s_wait_tensorcnt(0);
  __syncthreads();
  for (int kt = 0; kt < nK; ++kt) {
    const int buf = kt & 1;
    if (kt + 1 < nK) {
      stageA(kt + 1, buf ^ 1);                 // async DMA overlaps WMMAs below
      if (wid == 0) stageB(kt + 1, buf ^ 1);   // TDM overlaps WMMAs below
    }
    compute(buf);
    wait_async0();
    if (wid == 0) __builtin_amdgcn_s_wait_tensorcnt(0);
    __syncthreads();
  }

  // epilogue: bias (+GELU), write f32 and/or bf16
  #pragma unroll
  for (int ti = 0; ti < 2; ++ti) {
    #pragma unroll
    for (int tj = 0; tj < 4; ++tj) {
      v8f a = acc[ti][tj];
      int n = bn + wn + tj * 16 + l15;
      float bv = bias[n];
      #pragma unroll
      for (int r = 0; r < 8; ++r) {
        int m = bm + wm + ti * 16 + r + half * 8;
        float v = a[r] + bv;
        if (do_gelu) v = gelu_tanh(v);
        size_t o = (size_t)m * N + n;
        if (Cf) Cf[o] = v;
        if (Cb) Cb[o] = f2bf(v);
      }
    }
  }
}

// ---------------------------------------------------------------- sparse attention
// one wave per (b,h,nb); 7 gathered key blocks (16 keys each)
__global__ __launch_bounds__(64) void attn_sparse(const u16* __restrict__ qkv,
                                                  const int* __restrict__ kidx,
                                                  u16* __restrict__ aout) {
  __shared__ u16 Vt[2][64][128];   // [wave][dh][key] transposed V
  __shared__ u16 Pl[2][16][128];   // [wave][q][key] probs (padded to 128)
  const int wid  = threadIdx.x >> 5;
  const int lane = threadIdx.x & 31;
  const int half = lane >> 4, l15 = lane & 15;
  const int task = blockIdx.x * 2 + wid;          // 0..6143
  const int nb = task & (NBsz - 1);
  const int bh = task >> 8;
  const int b = bh / Hsz, h = bh % Hsz;

  // Q A-frags (direct from global, contiguous dh)
  Frag af[2];
  const u16* qb = qkv + (size_t)(b * Ssz + nb * 16 + l15) * QKVW + h * DHsz;
  #pragma unroll
  for (int ks = 0; ks < 2; ++ks) {
    af[ks].q[0] = *(const uint4*)(qb + ks * 32 + half * 8);
    af[ks].q[1] = *(const uint4*)(qb + ks * 32 + half * 8 + 16);
  }

  // S = Q @ K^T  (7 tiles of 16 keys; tile t == gathered block t)
  v8f sacc[7];
  #pragma unroll
  for (int t = 0; t < 7; ++t) {
    int kb = kidx[nb * 7 + t];
    const u16* kr = qkv + (size_t)(b * Ssz + kb * 16 + l15) * QKVW + Dsz + h * DHsz;
    v8f c = v8f_zero();
    #pragma unroll
    for (int ks = 0; ks < 2; ++ks) {
      Frag bf;
      bf.q[0] = *(const uint4*)(kr + ks * 32 + half * 8);
      bf.q[1] = *(const uint4*)(kr + ks * 32 + half * 8 + 16);
      c = wmma_bf16(af[ks], bf, c);
    }
    sacc[t] = c;
  }

  // stage V transposed: Vt[dh][key]; each lane covers 2 dh per key
  for (int key = 0; key < 112; ++key) {
    int kb = kidx[nb * 7 + (key >> 4)];
    const u16* vr = qkv + (size_t)(b * Ssz + kb * 16 + (key & 15)) * QKVW
                    + 2 * Dsz + h * DHsz + lane * 2;
    u32 vv = *(const u32*)vr;
    Vt[wid][lane * 2][key]     = (u16)(vv & 0xffffu);
    Vt[wid][lane * 2 + 1][key] = (u16)(vv >> 16);
  }
  for (int key = 112; key < 128; ++key) {
    Vt[wid][lane * 2][key] = 0; Vt[wid][lane * 2 + 1][key] = 0;
  }

  // softmax over 112 keys; lane stats cover rows r + half*8
  float rinv[8];
  #pragma unroll
  for (int r = 0; r < 8; ++r) {
    float sv[7];
    float mx = -1e30f;
    #pragma unroll
    for (int t = 0; t < 7; ++t) { sv[t] = sacc[t][r] * SCALE; mx = fmaxf(mx, sv[t]); }
    #pragma unroll
    for (int off = 1; off < 16; off <<= 1) mx = fmaxf(mx, __shfl_xor(mx, off, 16));
    float sum = 0.f;
    #pragma unroll
    for (int t = 0; t < 7; ++t) {
      float p = __expf(sv[t] - mx);
      sum += p;
      Pl[wid][r + half * 8][t * 16 + l15] = f2bf(p);
    }
    Pl[wid][r + half * 8][112 + l15] = 0;   // pad keys 112..127
    #pragma unroll
    for (int off = 1; off < 16; off <<= 1) sum += __shfl_xor(sum, off, 16);
    rinv[r] = 1.0f / sum;
  }
  __syncthreads();

  // O = P @ V  (K=128 padded, N=64)
  v8f acco[4];
  #pragma unroll
  for (int tj = 0; tj < 4; ++tj) acco[tj] = v8f_zero();
  #pragma unroll
  for (int ks = 0; ks < 4; ++ks) {
    Frag ap;
    ap.q[0] = *(const uint4*)&Pl[wid][l15][ks * 32 + half * 8];
    ap.q[1] = *(const uint4*)&Pl[wid][l15][ks * 32 + half * 8 + 16];
    #pragma unroll
    for (int tj = 0; tj < 4; ++tj) {
      Frag bv;
      bv.q[0] = *(const uint4*)&Vt[wid][tj * 16 + l15][ks * 32 + half * 8];
      bv.q[1] = *(const uint4*)&Vt[wid][tj * 16 + l15][ks * 32 + half * 8 + 16];
      acco[tj] = wmma_bf16(ap, bv, acco[tj]);
    }
  }

  // store to a[b, s, h*64+dh] bf16
  #pragma unroll
  for (int tj = 0; tj < 4; ++tj) {
    #pragma unroll
    for (int r = 0; r < 8; ++r) {
      int row = r + half * 8;
      size_t o = (size_t)(b * Ssz + nb * 16 + row) * Dsz + h * DHsz + tj * 16 + l15;
      aout[o] = f2bf(acco[tj][r] * rinv[r]);
    }
  }
}

// ---------------------------------------------------------------- edge blocks: full attention (flash-style)
__global__ __launch_bounds__(32) void attn_edge(const u16* __restrict__ qkv,
                                                u16* __restrict__ aout) {
  __shared__ u16 Vt[64][128];
  __shared__ u16 Pl[16][128];
  const int lane = threadIdx.x;
  const int half = lane >> 4, l15 = lane & 15;
  const int task = blockIdx.x;       // 0..47
  const int g = task & 1;
  const int bh = task >> 1;
  const int b = bh / Hsz, h = bh % Hsz;
  const int nb = g ? (NBsz - 1) : 0;

  Frag af[2];
  const u16* qb = qkv + (size_t)(b * Ssz + nb * 16 + l15) * QKVW + h * DHsz;
  #pragma unroll
  for (int ks = 0; ks < 2; ++ks) {
    af[ks].q[0] = *(const uint4*)(qb + ks * 32 + half * 8);
    af[ks].q[1] = *(const uint4*)(qb + ks * 32 + half * 8 + 16);
  }

  float mrow[8], lrow[8];
  v8f acco[4];
  #pragma unroll
  for (int r = 0; r < 8; ++r) { mrow[r] = -1e30f; lrow[r] = 0.f; }
  #pragma unroll
  for (int tj = 0; tj < 4; ++tj) acco[tj] = v8f_zero();

  for (int c = 0; c < Ssz / 128; ++c) {
    // stage V chunk transposed
    for (int key = 0; key < 128; ++key) {
      int s = c * 128 + key;
      const u16* vr = qkv + (size_t)(b * Ssz + s) * QKVW + 2 * Dsz + h * DHsz + lane * 2;
      u32 vv = *(const u32*)vr;
      Vt[lane * 2][key]     = (u16)(vv & 0xffffu);
      Vt[lane * 2 + 1][key] = (u16)(vv >> 16);
    }
    __syncthreads();

    // S chunk: 8 tiles of 16 keys
    v8f sacc[8];
    #pragma unroll
    for (int t = 0; t < 8; ++t) {
      const u16* kr = qkv + (size_t)(b * Ssz + c * 128 + t * 16 + l15) * QKVW
                      + Dsz + h * DHsz;
      v8f cc = v8f_zero();
      #pragma unroll
      for (int ks = 0; ks < 2; ++ks) {
        Frag bf;
        bf.q[0] = *(const uint4*)(kr + ks * 32 + half * 8);
        bf.q[1] = *(const uint4*)(kr + ks * 32 + half * 8 + 16);
        cc = wmma_bf16(af[ks], bf, cc);
      }
      sacc[t] = cc;
    }

    // streaming softmax update
    #pragma unroll
    for (int r = 0; r < 8; ++r) {
      float sv[8];
      float cm = -1e30f;
      #pragma unroll
      for (int t = 0; t < 8; ++t) { sv[t] = sacc[t][r] * SCALE; cm = fmaxf(cm, sv[t]); }
      #pragma unroll
      for (int off = 1; off < 16; off <<= 1) cm = fmaxf(cm, __shfl_xor(cm, off, 16));
      float mnew = fmaxf(mrow[r], cm);
      float alpha = __expf(mrow[r] - mnew);
      float cs = 0.f;
      #pragma unroll
      for (int t = 0; t < 8; ++t) {
        float p = __expf(sv[t] - mnew);
        cs += p;
        Pl[r + half * 8][t * 16 + l15] = f2bf(p);
      }
      #pragma unroll
      for (int off = 1; off < 16; off <<= 1) cs += __shfl_xor(cs, off, 16);
      lrow[r] = lrow[r] * alpha + cs;
      mrow[r] = mnew;
      #pragma unroll
      for (int tj = 0; tj < 4; ++tj) acco[tj][r] = acco[tj][r] * alpha;
    }
    __syncthreads();

    // O += P @ V
    #pragma unroll
    for (int ks = 0; ks < 4; ++ks) {
      Frag ap;
      ap.q[0] = *(const uint4*)&Pl[l15][ks * 32 + half * 8];
      ap.q[1] = *(const uint4*)&Pl[l15][ks * 32 + half * 8 + 16];
      #pragma unroll
      for (int tj = 0; tj < 4; ++tj) {
        Frag bv;
        bv.q[0] = *(const uint4*)&Vt[tj * 16 + l15][ks * 32 + half * 8];
        bv.q[1] = *(const uint4*)&Vt[tj * 16 + l15][ks * 32 + half * 8 + 16];
        acco[tj] = wmma_bf16(ap, bv, acco[tj]);
      }
    }
    __syncthreads();
  }

  #pragma unroll
  for (int tj = 0; tj < 4; ++tj) {
    #pragma unroll
    for (int r = 0; r < 8; ++r) {
      int row = r + half * 8;
      size_t o = (size_t)(b * Ssz + nb * 16 + row) * Dsz + h * DHsz + tj * 16 + l15;
      aout[o] = f2bf(acco[tj][r] / lrow[r]);
    }
  }
}

// ---------------------------------------------------------------- residual + LayerNorm
__global__ __launch_bounds__(256) void add_ln(float* __restrict__ x,
                                              const float* __restrict__ y,
                                              const float* __restrict__ g,
                                              const float* __restrict__ bt,
                                              u16* __restrict__ xb) {
  __shared__ float s1[256], s2[256];
  const int row = blockIdx.x, t = threadIdx.x;
  float v[3], a = 0.f, sq = 0.f;
  #pragma unroll
  for (int i = 0; i < 3; ++i) {
    int c = t + i * 256;
    float val = x[(size_t)row * Dsz + c] + y[(size_t)row * Dsz + c];
    v[i] = val; a += val; sq += val * val;
  }
  s1[t] = a; s2[t] = sq;
  __syncthreads();
  for (int off = 128; off > 0; off >>= 1) {
    if (t < off) { s1[t] += s1[t + off]; s2[t] += s2[t + off]; }
    __syncthreads();
  }
  float mean = s1[0] * (1.0f / Dsz);
  float var  = s2[0] * (1.0f / Dsz) - mean * mean;
  float inv  = rsqrtf(var + 1e-12f);
  #pragma unroll
  for (int i = 0; i < 3; ++i) {
    int c = t + i * 256;
    float o = (v[i] - mean) * inv * g[c] + bt[c];
    x[(size_t)row * Dsz + c]  = o;
    xb[(size_t)row * Dsz + c] = f2bf(o);
  }
}

// ---------------------------------------------------------------- heads
__global__ __launch_bounds__(256) void pool_kernel(const float* __restrict__ x,
                                                   const float* __restrict__ Wp,
                                                   const float* __restrict__ bp,
                                                   float* __restrict__ pool) {
  int idx = blockIdx.x * 256 + threadIdx.x;
  if (idx >= Bsz * Dsz) return;
  int bb = idx / Dsz, d = idx % Dsz;
  const float* xr = x + (size_t)bb * Ssz * Dsz;   // s == 0
  float s = bp[d];
  for (int k = 0; k < Dsz; ++k) s += xr[k] * Wp[(size_t)k * Dsz + d];
  pool[idx] = tanhf(s);
}

__global__ __launch_bounds__(256) void logits_kernel(const float* __restrict__ x,
                                                     const float* __restrict__ sw,
                                                     const float* __restrict__ ew,
                                                     float* __restrict__ ls,
                                                     float* __restrict__ le) {
  __shared__ float r1[256], r2[256];
  const int row = blockIdx.x, t = threadIdx.x;
  const float* xr = x + (size_t)row * Dsz;
  float a = 0.f, bv = 0.f;
  for (int i = t; i < Dsz; i += 256) { float xv = xr[i]; a += xv * sw[i]; bv += xv * ew[i]; }
  r1[t] = a; r2[t] = bv;
  __syncthreads();
  for (int off = 128; off > 0; off >>= 1) {
    if (t < off) { r1[t] += r1[t + off]; r2[t] += r2[t + off]; }
    __syncthreads();
  }
  if (t == 0) { ls[row] = r1[0]; le[row] = r2[0]; }
}

__global__ __launch_bounds__(256) void softmax4096(const float* __restrict__ ls,
                                                   const float* __restrict__ le,
                                                   float* __restrict__ out) {
  __shared__ float red[256];
  const int b = blockIdx.x & 1, which = blockIdx.x >> 1;
  const float* src = (which ? le : ls) + b * Ssz;
  float* dst = out + which * ROWS + b * Ssz;
  const int t = threadIdx.x;
  float v[16], mx = -1e30f;
  #pragma unroll
  for (int i = 0; i < 16; ++i) { v[i] = src[t + i * 256]; mx = fmaxf(mx, v[i]); }
  red[t] = mx; __syncthreads();
  for (int off = 128; off > 0; off >>= 1) {
    if (t < off) red[t] = fmaxf(red[t], red[t + off]);
    __syncthreads();
  }
  mx = red[0]; __syncthreads();
  float s = 0.f;
  #pragma unroll
  for (int i = 0; i < 16; ++i) { v[i] = __expf(v[i] - mx); s += v[i]; }
  red[t] = s; __syncthreads();
  for (int off = 128; off > 0; off >>= 1) {
    if (t < off) red[t] += red[t + off];
    __syncthreads();
  }
  float inv = 1.0f / red[0];
  #pragma unroll
  for (int i = 0; i < 16; ++i) dst[t + i * 256] = v[i] * inv;
}

__global__ __launch_bounds__(32) void disc_kernel(const float* __restrict__ pool,
                                                  const float* __restrict__ d1W,
                                                  const float* __restrict__ d1b,
                                                  const float* __restrict__ d2W,
                                                  const float* __restrict__ d2b,
                                                  float* __restrict__ dd) {
  __shared__ float hsh[20];
  const int t = threadIdx.x;
  if (t < 20) {
    float s = d1b[t];
    for (int k = 0; k < Dsz; ++k) s += pool[k] * d1W[k * 20 + t];
    hsh[t] = s;
  }
  __syncthreads();
  if (t == 0) {
    float z0 = d2b[0], z1 = d2b[1];
    for (int j = 0; j < 20; ++j) { z0 += hsh[j] * d2W[j * 2]; z1 += hsh[j] * d2W[j * 2 + 1]; }
    float m = fmaxf(z0, z1);
    float e0 = __expf(z0 - m), e1 = __expf(z1 - m);
    float s = e0 + e1;
    dd[0] = e0 / s; dd[1] = e1 / s;
  }
}

// ---------------------------------------------------------------- launch
extern "C" void kernel_launch(void* const* d_in, const int* in_sizes, int n_in,
                              void* d_out, int out_size, void* d_ws, size_t ws_size,
                              hipStream_t stream) {
  const int*   input_ids = (const int*)d_in[0];
  const int*   key_idx   = (const int*)d_in[1];
  const float* embed     = (const float*)d_in[2];
  const float* Wqkv      = (const float*)d_in[3];
  const float* bqkv      = (const float*)d_in[4];
  const float* Wo        = (const float*)d_in[5];
  const float* bo        = (const float*)d_in[6];
  const float* ln1_g     = (const float*)d_in[7];
  const float* ln1_b     = (const float*)d_in[8];
  const float* Wff1      = (const float*)d_in[9];
  const float* bff1      = (const float*)d_in[10];
  const float* Wff2      = (const float*)d_in[11];
  const float* bff2      = (const float*)d_in[12];
  const float* ln2_g     = (const float*)d_in[13];
  const float* ln2_b     = (const float*)d_in[14];
  const float* Wp        = (const float*)d_in[15];
  const float* bp        = (const float*)d_in[16];
  const float* start_w   = (const float*)d_in[17];
  const float* end_w     = (const float*)d_in[18];
  const float* d1_W      = (const float*)d_in[19];
  const float* d1_b      = (const float*)d_in[20];
  const float* d2_W      = (const float*)d_in[21];
  const float* d2_b      = (const float*)d_in[22];
  float* out = (float*)d_out;
  char*  ws  = (char*)d_ws;

  // workspace layout (all chunk sizes are 256B multiples)
  size_t o_xf   = 0;
  size_t o_xb   = o_xf   + (size_t)ROWS * Dsz * 4;
  size_t o_qkv  = o_xb   + (size_t)ROWS * Dsz * 2;
  size_t o_ab   = o_qkv  + (size_t)ROWS * QKVW * 2;
  size_t o_y    = o_ab   + (size_t)ROWS * Dsz * 2;
  size_t o_h    = o_y    + (size_t)ROWS * Dsz * 4;
  size_t o_wqkv = o_h    + (size_t)ROWS * FFsz * 2;
  size_t o_wo   = o_wqkv + (size_t)2 * Dsz * QKVW * 2;
  size_t o_w1   = o_wo   + (size_t)2 * Dsz * Dsz * 2;
  size_t o_w2   = o_w1   + (size_t)2 * Dsz * FFsz * 2;
  size_t o_pool = o_w2   + (size_t)2 * FFsz * Dsz * 2;
  size_t o_ls   = o_pool + (size_t)Bsz * Dsz * 4;
  size_t o_le   = o_ls   + (size_t)ROWS * 4;

  float* xf    = (float*)(ws + o_xf);
  u16*   xb    = (u16*)  (ws + o_xb);
  u16*   qkvb  = (u16*)  (ws + o_qkv);
  u16*   ab    = (u16*)  (ws + o_ab);
  float* yf    = (float*)(ws + o_y);
  u16*   hb    = (u16*)  (ws + o_h);
  u16*   wqkvt = (u16*)  (ws + o_wqkv);   // (L, 2304, 768)
  u16*   wot   = (u16*)  (ws + o_wo);     // (L, 768, 768)
  u16*   w1t   = (u16*)  (ws + o_w1);     // (L, 3072, 768)
  u16*   w2t   = (u16*)  (ws + o_w2);     // (L, 768, 3072)
  float* poolf = (float*)(ws + o_pool);
  float* lsf   = (float*)(ws + o_ls);
  float* lef   = (float*)(ws + o_le);

  // one-time: weights f32 (K,N) -> bf16 transposed (N,K); L2-resident afterwards
  transpose_f2bf<<<dim3(QKVW / 32, Dsz  / 32, 2), 256, 0, stream>>>(Wqkv, wqkvt, Dsz,  QKVW);
  transpose_f2bf<<<dim3(Dsz  / 32, Dsz  / 32, 2), 256, 0, stream>>>(Wo,   wot,   Dsz,  Dsz);
  transpose_f2bf<<<dim3(FFsz / 32, Dsz  / 32, 2), 256, 0, stream>>>(Wff1, w1t,   Dsz,  FFsz);
  transpose_f2bf<<<dim3(Dsz  / 32, FFsz / 32, 2), 256, 0, stream>>>(Wff2, w2t,   FFsz, Dsz);

  embed_kernel<<<ROWS, 256, 0, stream>>>(input_ids, embed, xf, xb);

  for (int l = 0; l < 2; ++l) {
    // qkv = x @ Wqkv + bqkv  (bf16 out)
    gemm_bf16<<<dim3(QKVW / 128, ROWS / 128), 256, 0, stream>>>(
        xb, wqkvt + (size_t)l * Dsz * QKVW, bqkv + l * QKVW,
        nullptr, qkvb, QKVW, Dsz, 0);
    // sparse block attention + full-attention edge blocks
    attn_sparse<<<(Bsz * Hsz * NBsz) / 2, 64, 0, stream>>>(qkvb, key_idx, ab);
    attn_edge<<<Bsz * Hsz * 2, 32, 0, stream>>>(qkvb, ab);
    // y = a @ Wo + bo  (fp32 out)
    gemm_bf16<<<dim3(Dsz / 128, ROWS / 128), 256, 0, stream>>>(
        ab, wot + (size_t)l * Dsz * Dsz, bo + l * Dsz,
        yf, nullptr, Dsz, Dsz, 0);
    add_ln<<<ROWS, 256, 0, stream>>>(xf, yf, ln1_g + l * Dsz, ln1_b + l * Dsz, xb);
    // h = gelu(x @ Wff1 + bff1)  (bf16 out)
    gemm_bf16<<<dim3(FFsz / 128, ROWS / 128), 256, 0, stream>>>(
        xb, w1t + (size_t)l * Dsz * FFsz, bff1 + l * FFsz,
        nullptr, hb, FFsz, Dsz, 1);
    // y = h @ Wff2 + bff2  (fp32 out)
    gemm_bf16<<<dim3(Dsz / 128, ROWS / 128), 256, 0, stream>>>(
        hb, w2t + (size_t)l * FFsz * Dsz, bff2 + l * Dsz,
        yf, nullptr, Dsz, FFsz, 0);
    add_ln<<<ROWS, 256, 0, stream>>>(xf, yf, ln2_g + l * Dsz, ln2_b + l * Dsz, xb);
  }

  pool_kernel<<<(Bsz * Dsz + 255) / 256, 256, 0, stream>>>(xf, Wp, bp, poolf);
  logits_kernel<<<ROWS, 256, 0, stream>>>(xf, start_w, end_w, lsf, lef);
  softmax4096<<<4, 256, 0, stream>>>(lsf, lef, out);
  disc_kernel<<<1, 32, 0, stream>>>(poolf, d1_W, d1_b, d2_W, d2_b, out + 2 * ROWS);
}
